// Network_9191230014067
// MI455X (gfx1250) — matrix operand order repaired and checked by reading, not compiled, not run on hardware
//
#include <hip/hip_runtime.h>
#include <hip/hip_bf16.h>

// Problem constants (from reference)
#define POSDIM 128
#define MID    512
#define RANK   1024
#define NPTS   320
#define TWO_PI 6.283185307179586f

typedef __attribute__((ext_vector_type(16))) __bf16 v16bf;
typedef __attribute__((ext_vector_type(8)))  float  v8f;
typedef __hip_bfloat16 bf16_t;

union FragBF { uint4 q[2]; v16bf v; unsigned u[8]; };

// ---------------------------------------------------------------------------
// WMMA fragment loaders per CDNA5 ISA 7.12.2 VGPR layouts (wave32).
// A (16x32 bf16, MxK): lanes 0-15 -> M=lane, K chunks [k0+8*hi .. +7] and
// [k0+16+8*hi .. +7] where hi = lane>>4.
// ---------------------------------------------------------------------------
__device__ __forceinline__ v16bf load_a_frag(const bf16_t* A, int lda,
                                             int row0, int k0, int lane) {
  const int m = lane & 15, hi = lane >> 4;
  const bf16_t* p = A + (size_t)(row0 + m) * lda + k0 + hi * 8;
  FragBF u;
  u.q[0] = *(const uint4*)p;        // K = k0 + 8*hi + (0..7)
  u.q[1] = *(const uint4*)(p + 16); // K = k0 + 16 + 8*hi + (0..7)
  return u.v;
}

// B (32x16 bf16, KxN) with B[r][n] = Wrow-major[n][r]: lane holds column
// n = lane&15, K = k0 + 8*hi + (0..7) and k0 + 16 + 8*hi + (0..7) of W's row n.
__device__ __forceinline__ v16bf load_b_frag(const bf16_t* Wm, int ldw,
                                             int col0, int k0, int lane) {
  const int n = lane & 15, hi = lane >> 4;
  const bf16_t* p = Wm + (size_t)(col0 + n) * ldw + k0 + hi * 8;
  FragBF u;
  u.q[0] = *(const uint4*)p;
  u.q[1] = *(const uint4*)(p + 16);
  return u.v;
}

// Broadcast row of V in A-fragment K-layout (same K positions for every M).
__device__ __forceinline__ v16bf load_vrow_frag(const bf16_t* Vrow, int k0, int lane) {
  const int hi = lane >> 4;
  const bf16_t* p = Vrow + k0 + hi * 8;
  FragBF u;
  u.q[0] = *(const uint4*)p;
  u.q[1] = *(const uint4*)(p + 16);
  return u.v;
}

// Elementwise bf16 multiply of two fragments via v_pk_mul_bf16 (VOP3P).
// Not volatile: pure computation, let the scheduler interleave with WMMA
// (16-bit WMMA co-executes with normal VALU per ISA 8.x TRANS tracking).
__device__ __forceinline__ v16bf pk_mul16(v16bf a, v16bf b) {
  FragBF ua, ub, ud;
  ua.v = a; ub.v = b;
#pragma unroll
  for (int i = 0; i < 8; ++i)
    asm("v_pk_mul_bf16 %0, %1, %2"
        : "=v"(ud.u[i]) : "v"(ua.u[i]), "v"(ub.u[i]));
  return ud.v;
}

__device__ __forceinline__ v8f wmma_bf16(v16bf a, v16bf b, v8f c) {
  return __builtin_amdgcn_wmma_f32_16x16x32_bf16(
      /*neg_a=*/false, a, /*neg_b=*/false, b,
      /*c_mod=*/(short)0, c, /*reuse_a=*/false, /*reuse_b=*/false);
}

// ---------------------------------------------------------------------------
// f32 -> bf16 conversion (weights), RNE via HIP helper.
// ---------------------------------------------------------------------------
__global__ void f32_to_bf16_kernel(const float* __restrict__ src,
                                   bf16_t* __restrict__ dst, int n) {
  int i = blockIdx.x * blockDim.x + threadIdx.x;
  if (i < n) dst[i] = __float2bfloat16(src[i]);
}

// ---------------------------------------------------------------------------
// Normalize coords per tower + random Fourier features -> bf16 (3,320,128).
// One block per tower, 320 threads.
// ---------------------------------------------------------------------------
__global__ void encode_kernel(const float* __restrict__ coords,
                              const float* __restrict__ b_enc,
                              bf16_t* __restrict__ enc) {
  const int t = blockIdx.x;
  const int n = threadIdx.x;        // 0..319
  __shared__ float sv[NPTS];
  __shared__ float s_min, s_max;
  const float c = coords[(size_t)t * NPTS + n];
  sv[n] = c;
  __syncthreads();
  if (n == 0) {
    float mn = sv[0], mx = sv[0];
    for (int i = 1; i < NPTS; ++i) { mn = fminf(mn, sv[i]); mx = fmaxf(mx, sv[i]); }
    s_min = mn; s_max = mx;
  }
  __syncthreads();
  const float mn = s_min, den = s_max - s_min;
  const float x = (den == 0.0f) ? 0.0f : (c - mn) / den;
  bf16_t* e = enc + ((size_t)t * NPTS + n) * POSDIM;
#pragma unroll 4
  for (int q = 0; q < POSDIM / 2; ++q) {
    const float v = TWO_PI * x * b_enc[q];
    e[q]              = __float2bfloat16(__cosf(v));
    e[POSDIM / 2 + q] = __float2bfloat16(__sinf(v));
  }
}

// ---------------------------------------------------------------------------
// Generic per-tower MLP layer: O = act(A @ W^T + bias)
//   A: (M x K) bf16 row-major     (per-tower stride sA elements)
//   W: (N x K) bf16 row-major     (per-tower stride sW)
//   O: (M x N) bf16               (per-tower stride sO)
//   Of32: optional f32 copy of O  (final layer -> U/V/W outputs)
// grid = (N/64, M/16, 3), block = 128 (4 waves, one 16x16 tile each).
// MLP stages total ~2 GFLOP: correctness of structure > tuning here.
// ---------------------------------------------------------------------------
__global__ void __launch_bounds__(128)
mlp_gemm_bf16(const bf16_t* __restrict__ A, long sA,
              const bf16_t* __restrict__ Wm, long sW,
              const float* __restrict__ bias, long sB,
              bf16_t* __restrict__ Obf, long sO,
              float* __restrict__ Of32, long sF,
              int M, int N, int K, int relu) {
  const int lane = threadIdx.x & 31;
  const int wave = threadIdx.x >> 5;
  const int t  = blockIdx.z;
  const int m0 = blockIdx.y * 16;
  const int n0 = blockIdx.x * 64 + wave * 16;
  const bf16_t* At = A  + (size_t)t * sA;
  const bf16_t* Wt = Wm + (size_t)t * sW;

  v8f acc = {};
  for (int k0 = 0; k0 < K; k0 += 32) {
    v16bf a = load_a_frag(At, K, m0, k0, lane);
    v16bf b = load_b_frag(Wt, K, n0, k0, lane);
    acc = wmma_bf16(a, b, acc);
  }

  const int col = n0 + (lane & 15);
  const int hi  = lane >> 4;
  const float bz = bias[(size_t)t * sB + col];
  bf16_t* Ot = Obf + (size_t)t * sO;
  float*  Ft = Of32 ? (Of32 + (size_t)t * sF) : nullptr;
#pragma unroll
  for (int r = 0; r < 8; ++r) {
    const int row = m0 + r + 8 * hi;       // C/D layout: VGPR r, half hi
    float val = acc[r] + bz;
    if (relu) val = fmaxf(val, 0.0f);
    Ot[(size_t)row * N + col] = __float2bfloat16(val);
    if (Ft) Ft[(size_t)row * N + col] = val;
  }
}

// ---------------------------------------------------------------------------
// out[i,j,k] = sum_r U[i,r] V[j,r] W[k,r]  -- the dominant ~67-GFLOP stage.
// For fixed j: out[j] = (U .* V[j,:]) @ W^T, M=N=320, K=1024.
// grid = (5, 5, 320): x -> k-tile(64), y -> i-tile(64), z -> j.
// block = 64 threads (2 waves); each wave owns a 32x64 macro-tile:
// 2 A-frags (V-scaled) x 4 B-frags -> 8 WMMAs per 32-K step against
// 14x16B loads -> 1.75 loads/WMMA (was 2.5 with 32x32 tiles).
// U/V/W bf16 total ~2MB -> fully L2-resident (192MB L2); compute-bound.
// ---------------------------------------------------------------------------
__global__ void __launch_bounds__(64)
cp_outer_gemm(const bf16_t* __restrict__ U, const bf16_t* __restrict__ V,
              const bf16_t* __restrict__ Wf, float* __restrict__ out) {
  const int lane = threadIdx.x & 31;
  const int wave = threadIdx.x >> 5;           // 0..1
  const int j  = blockIdx.z;
  const int i0 = blockIdx.y * 64 + wave * 32;  // 32 rows per wave
  const int k0 = blockIdx.x * 64;              // 64 cols per wave
  const bf16_t* Vrow = V + (size_t)j * RANK;

  v8f acc[2][4] = {};
  for (int r0 = 0; r0 < RANK; r0 += 32) {
    const v16bf vf = load_vrow_frag(Vrow, r0, lane);
    const v16bf a0 = pk_mul16(load_a_frag(U, RANK, i0,      r0, lane), vf);
    const v16bf a1 = pk_mul16(load_a_frag(U, RANK, i0 + 16, r0, lane), vf);
    v16bf b[4];
#pragma unroll
    for (int nb = 0; nb < 4; ++nb)
      b[nb] = load_b_frag(Wf, RANK, k0 + 16 * nb, r0, lane);
#pragma unroll
    for (int nb = 0; nb < 4; ++nb) {
      acc[0][nb] = wmma_bf16(a0, b[nb], acc[0][nb]);
      acc[1][nb] = wmma_bf16(a1, b[nb], acc[1][nb]);
    }
  }

  const int n = lane & 15, hi = lane >> 4;
#pragma unroll
  for (int mt = 0; mt < 2; ++mt) {
#pragma unroll
    for (int r = 0; r < 8; ++r) {
      const size_t base =
          (size_t)(i0 + 16 * mt + r + 8 * hi) * (NPTS * NPTS) + (size_t)j * NPTS;
#pragma unroll
      for (int nb = 0; nb < 4; ++nb)
        out[base + k0 + 16 * nb + n] = acc[mt][nb][r];
    }
  }
}

// ---------------------------------------------------------------------------
// Host-side orchestration.
// ---------------------------------------------------------------------------
extern "C" void kernel_launch(void* const* d_in, const int* in_sizes, int n_in,
                              void* d_out, int out_size, void* d_ws, size_t ws_size,
                              hipStream_t stream) {
  const float* coords = (const float*)d_in[0];
  const float* b_enc  = (const float*)d_in[1];
  const float* W0 = (const float*)d_in[2];
  const float* b0 = (const float*)d_in[3];
  const float* W1 = (const float*)d_in[4];
  const float* b1 = (const float*)d_in[5];
  const float* W2 = (const float*)d_in[6];
  const float* b2 = (const float*)d_in[7];
  const float* W3 = (const float*)d_in[8];
  const float* b3 = (const float*)d_in[9];

  float* out  = (float*)d_out;                              // (320,320,320)
  float* Uvw  = out + (size_t)NPTS * NPTS * NPTS;           // U,V,W f32, each 320x1024

  // Workspace layout (bf16 elements)
  const size_t cW0 = 3ull * MID  * POSDIM;   // 196608
  const size_t cW1 = 3ull * MID  * MID;      // 786432
  const size_t cW2 = cW1;
  const size_t cW3 = 3ull * RANK * MID;      // 1572864
  const size_t cEnc = 3ull * NPTS * POSDIM;  // 122880
  const size_t cH   = 3ull * NPTS * MID;     // 491520

  bf16_t* w0b = (bf16_t*)d_ws;
  bf16_t* w1b = w0b + cW0;
  bf16_t* w2b = w1b + cW1;
  bf16_t* w3b = w2b + cW2;
  bf16_t* enc = w3b + cW3;
  bf16_t* hA  = enc + cEnc;
  bf16_t* hB  = hA  + cH;
  bf16_t* fbf = hB  + cH;   // (3,320,1024) bf16 -> U,V,W for the big GEMM

  // 1) Convert weights to bf16.
  {
    const int T = 256;
    f32_to_bf16_kernel<<<(int)((cW0 + T - 1) / T), T, 0, stream>>>(W0, w0b, (int)cW0);
    f32_to_bf16_kernel<<<(int)((cW1 + T - 1) / T), T, 0, stream>>>(W1, w1b, (int)cW1);
    f32_to_bf16_kernel<<<(int)((cW2 + T - 1) / T), T, 0, stream>>>(W2, w2b, (int)cW2);
    f32_to_bf16_kernel<<<(int)((cW3 + T - 1) / T), T, 0, stream>>>(W3, w3b, (int)cW3);
  }

  // 2) Normalize + Fourier encode.
  encode_kernel<<<3, NPTS, 0, stream>>>(coords, b_enc, enc);

  // 3) MLP towers (4 WMMA GEMM layers, batched over towers via grid.z).
  // L0: (320x128)@(512x128)^T -> hA, relu
  mlp_gemm_bf16<<<dim3(MID / 64, NPTS / 16, 3), 128, 0, stream>>>(
      enc, (long)(NPTS * POSDIM), w0b, (long)(MID * POSDIM),
      b0, (long)MID, hA, (long)(NPTS * MID), nullptr, 0,
      NPTS, MID, POSDIM, 1);
  // L1: hA @ W1^T -> hB, relu
  mlp_gemm_bf16<<<dim3(MID / 64, NPTS / 16, 3), 128, 0, stream>>>(
      hA, (long)(NPTS * MID), w1b, (long)(MID * MID),
      b1, (long)MID, hB, (long)(NPTS * MID), nullptr, 0,
      NPTS, MID, MID, 1);
  // L2: hB @ W2^T -> hA, relu
  mlp_gemm_bf16<<<dim3(MID / 64, NPTS / 16, 3), 128, 0, stream>>>(
      hB, (long)(NPTS * MID), w2b, (long)(MID * MID),
      b2, (long)MID, hA, (long)(NPTS * MID), nullptr, 0,
      NPTS, MID, MID, 1);
  // L3: hA @ W3^T -> fbf (bf16) and Uvw (f32 outputs U,V,W), no relu
  mlp_gemm_bf16<<<dim3(RANK / 64, NPTS / 16, 3), 128, 0, stream>>>(
      hA, (long)(NPTS * MID), w3b, (long)(RANK * MID),
      b3, (long)RANK, fbf, (long)(NPTS * RANK), Uvw, (long)(NPTS * RANK),
      NPTS, RANK, MID, 0);

  // 4) out[i,j,k] = sum_r U[i,r] V[j,r] W[k,r]  (dominant stage).
  const bf16_t* Ub = fbf;
  const bf16_t* Vb = fbf + (size_t)NPTS * RANK;
  const bf16_t* Wb = fbf + 2ull * NPTS * RANK;
  cp_outer_gemm<<<dim3(NPTS / 64, NPTS / 64, NPTS), 64, 0, stream>>>(Ub, Vb, Wb, out);
}
